// Attention_61108794687653
// MI455X (gfx1250) — matrix operand order repaired and checked by reading, not compiled, not run on hardware
//
#include <hip/hip_runtime.h>
#include <hip/hip_bf16.h>

// ---------------------------------------------------------------------------
// Types for CDNA5 WMMA f16 path
// ---------------------------------------------------------------------------
typedef _Float16 v16h __attribute__((ext_vector_type(16)));
typedef _Float16 v8h  __attribute__((ext_vector_type(8)));
typedef float    v8f  __attribute__((ext_vector_type(8)));

#define DIM 1024
#define BQ  8
#define TI  2048
#define TJ  2048

#define WMMA16(a, b, c) __builtin_amdgcn_wmma_f32_16x16x32_f16( \
    false, (a), false, (b), (short)0, (c), false, false)

// ---------------------------------------------------------------------------
// Fragment loaders (wave32, V_WMMA_F32_16X16X32_F16 layouts per ISA 7.12.2)
//
// A (16x32 f16): lane L holds row M = L&15.
//   halves 0..7 -> K = kb+0..7, halves 8..15 -> K = kb+16..23, kb = 8*(L>=16).
// B (32x16 f16): lane L holds column N = L&15.
//   halves 0..7 -> K = kb+0..7, halves 8..15 -> K = kb+16..23, kb = 8*(L>=16).
//   All B operands are stored "one source row per output column", so each
//   lane reads 2 x b128.
// C/D (16x16 f32): lane L holds column N = L&15; element r -> row
//   M = r + 8*(L>=16).
// ---------------------------------------------------------------------------
__device__ __forceinline__ v16h load_a16(const _Float16* __restrict__ p, int ld, int lane) {
    int r  = lane & 15;
    int kb = (lane >> 4) << 3;
    const _Float16* b0 = p + (size_t)r * ld + kb;
    union { v16h v; v8h h[2]; } u;
    u.h[0] = *(const v8h*)(b0);
    u.h[1] = *(const v8h*)(b0 + 16);
    return u.v;
}

__device__ __forceinline__ v16h load_b16(const _Float16* __restrict__ p, int ld, int lane) {
    int n  = lane & 15;
    int kb = (lane >> 4) << 3;
    const _Float16* b0 = p + (size_t)n * ld + kb;
    union { v16h v; v8h h[2]; } u;
    u.h[0] = *(const v8h*)(b0);
    u.h[1] = *(const v8h*)(b0 + 16);
    return u.v;
}

// ---------------------------------------------------------------------------
// Kernel 1: fused LayerNorm (f32) -> f16 cast.  One 256-thread block per row.
// ---------------------------------------------------------------------------
__global__ void __launch_bounds__(256)
ln_cast_kernel(const float* __restrict__ x, const float* __restrict__ g,
               const float* __restrict__ b, _Float16* __restrict__ y) {
    __shared__ float sred[16];
    const int row = blockIdx.x;
    const float* xr = x + (size_t)row * DIM;

    float v[4], s = 0.f, ss = 0.f;
#pragma unroll
    for (int i = 0; i < 4; ++i) {
        v[i] = xr[threadIdx.x + i * 256];
        s  += v[i];
        ss += v[i] * v[i];
    }
#pragma unroll
    for (int m = 1; m < 32; m <<= 1) {
        s  += __shfl_xor(s,  m, 32);
        ss += __shfl_xor(ss, m, 32);
    }
    const int wave = threadIdx.x >> 5;
    if ((threadIdx.x & 31) == 0) { sred[wave] = s; sred[8 + wave] = ss; }
    __syncthreads();
    s = 0.f; ss = 0.f;
#pragma unroll
    for (int w = 0; w < 8; ++w) { s += sred[w]; ss += sred[8 + w]; }

    const float mu  = s  * (1.0f / DIM);
    const float var = ss * (1.0f / DIM) - mu * mu;
    const float rs  = rsqrtf(var + 1e-5f);

    _Float16* yr = y + (size_t)row * DIM;
#pragma unroll
    for (int i = 0; i < 4; ++i) {
        int c = threadIdx.x + i * 256;
        yr[c] = (_Float16)((v[i] - mu) * rs * g[c] + b[c]);
    }
}

// ---------------------------------------------------------------------------
// Kernel 2: f32 -> f16 weight conversion
// ---------------------------------------------------------------------------
__global__ void __launch_bounds__(256)
f32_to_f16_kernel(const float* __restrict__ in, _Float16* __restrict__ out, int n) {
    int i = blockIdx.x * 256 + threadIdx.x;
    if (i < n) out[i] = (_Float16)in[i];
}

// ---------------------------------------------------------------------------
// Kernel 3: projection GEMM  Y = X @ W^T + bias (torch Linear semantics)
//   X: (16384,1024) f16   W: (1024,1024) f16 W[e][d]   bias f32
//   Each wave computes a 32x128 strip: 2 row tiles x 8 col tiles; every W
//   B-fragment feeds TWO WMMAs.  minWavesPerEU=1 releases VGPRs so the
//   scheduler can hold several B fragments in flight (kills wait-0 drains).
//   transposeV==0: Y f16 row-major (Q,K). transposeV==1: Vt[b][d][j]=V[b][j][d].
// ---------------------------------------------------------------------------
__global__ void __launch_bounds__(256, 1)
proj_gemm_kernel(const _Float16* __restrict__ X, const _Float16* __restrict__ W,
                 const float* __restrict__ bias, _Float16* __restrict__ Y,
                 int transposeV) {
    const int lane = threadIdx.x & 31;
    const int wave = threadIdx.x >> 5;
    const int row0 = blockIdx.x * 32;    // 0..16352
    const int col0 = wave * 128;         // 0..896

    v8f acc[2][8] = {};
    for (int k0 = 0; k0 < DIM; k0 += 32) {
        v16h a0 = load_a16(X + (size_t)row0 * DIM + k0, DIM, lane);
        v16h a1 = load_a16(X + (size_t)(row0 + 16) * DIM + k0, DIM, lane);
#pragma unroll
        for (int t = 0; t < 8; t += 2) {
            v16h bf0 = load_b16(W + (size_t)(col0 + t * 16) * DIM + k0, DIM, lane);
            v16h bf1 = load_b16(W + (size_t)(col0 + (t + 1) * 16) * DIM + k0, DIM, lane);
            acc[0][t]     = WMMA16(a0, bf0, acc[0][t]);
            acc[1][t]     = WMMA16(a1, bf0, acc[1][t]);
            acc[0][t + 1] = WMMA16(a0, bf1, acc[0][t + 1]);
            acc[1][t + 1] = WMMA16(a1, bf1, acc[1][t + 1]);
        }
    }

    const int cbl  = lane & 15;
    const int half = lane >> 4;
    if (!transposeV) {
#pragma unroll
        for (int t = 0; t < 8; ++t) {
            int col = col0 + t * 16 + cbl;
            float bv = bias[col];
#pragma unroll
            for (int qt = 0; qt < 2; ++qt) {
                int rbase = row0 + qt * 16 + half * 8;
#pragma unroll
                for (int r = 0; r < 8; ++r)
                    Y[(size_t)(rbase + r) * DIM + col] = (_Float16)(acc[qt][t][r] + bv);
            }
        }
    } else {
        const int b  = row0 >> 11;
        const int jb = (row0 & (TJ - 1)) + half * 8;
#pragma unroll
        for (int t = 0; t < 8; ++t) {
            int col = col0 + t * 16 + cbl;            // d
            float bv = bias[col];
#pragma unroll
            for (int qt = 0; qt < 2; ++qt) {
                int j = jb + qt * 16;
#pragma unroll
                for (int r = 0; r < 8; ++r)
                    Y[((size_t)b * DIM + col) * TJ + (j + r)] = (_Float16)(acc[qt][t][r] + bv);
            }
        }
    }
}

// ---------------------------------------------------------------------------
// Kernel 4: attention. One WG (16 waves, 512 thr) per (batch, 32-query tile).
//   Phase 1: wave w computes S[32][128] slab (cols w*128..) in WMMA regs;
//            each K B-fragment feeds 2 WMMAs.
//   Phase 2: exact softmax across full 2048 row: half-wave shuffles +
//            two-stage LDS reduction; P compressed to f16 in registers.
//   Phase 3: P staged in LDS in two 1024-col chunks (64KB tile), O = P@V
//            from transposed Vt; each Vt B-fragment feeds 2 WMMAs.
//   minWavesPerEU=1: 16 resident waves (4/SIMD) already saturate the matrix
//   pipes; spend VGPRs on deeper load pipelining instead of extra waves.
// ---------------------------------------------------------------------------
__global__ void __launch_bounds__(512, 1)
attn_kernel(const _Float16* __restrict__ Q, const _Float16* __restrict__ K,
            const _Float16* __restrict__ Vt, float* __restrict__ O) {
    __shared__ _Float16 Psh[32][1024];                  // 65536 bytes
    float* red  = reinterpret_cast<float*>(&Psh[0][0]); // [512] per-wave partials
    float* red2 = red + 512;                            // [32] final row stats

    const int lane  = threadIdx.x & 31;
    const int wave  = threadIdx.x >> 5;
    const int half  = lane >> 4;
    const int batch = blockIdx.x >> 6;
    const int q0    = (blockIdx.x & 63) * 32;

    const _Float16* Qb = Q  + ((size_t)batch * TI + q0) * DIM;
    const _Float16* Kb = K  + (size_t)batch * TJ * DIM;
    const _Float16* Vb = Vt + (size_t)batch * DIM * TJ;

    const int jw = wave * 128;                          // S column base

    // ---- Phase 1: S = (Q K^T) * scale, 32x128 per wave ----
    v8f acc[2][8] = {};
    for (int k0 = 0; k0 < DIM; k0 += 32) {
        v16h a0 = load_a16(Qb + k0, DIM, lane);
        v16h a1 = load_a16(Qb + (size_t)16 * DIM + k0, DIM, lane);
#pragma unroll
        for (int t = 0; t < 8; t += 2) {
            v16h bf0 = load_b16(Kb + (size_t)(jw + t * 16) * DIM + k0, DIM, lane);
            v16h bf1 = load_b16(Kb + (size_t)(jw + (t + 1) * 16) * DIM + k0, DIM, lane);
            acc[0][t]     = WMMA16(a0, bf0, acc[0][t]);
            acc[1][t]     = WMMA16(a1, bf0, acc[1][t]);
            acc[0][t + 1] = WMMA16(a0, bf1, acc[0][t + 1]);
            acc[1][t + 1] = WMMA16(a1, bf1, acc[1][t + 1]);
        }
    }

    // ---- Phase 2: softmax stats ----
    const float scale = 0.03125f;                       // DIM^-0.5
    float pmax[2][8];
#pragma unroll
    for (int qt = 0; qt < 2; ++qt)
#pragma unroll
        for (int r = 0; r < 8; ++r) pmax[qt][r] = -3.0e38f;
#pragma unroll
    for (int qt = 0; qt < 2; ++qt)
#pragma unroll
        for (int t = 0; t < 8; ++t)
#pragma unroll
            for (int r = 0; r < 8; ++r) {
                acc[qt][t][r] *= scale;
                pmax[qt][r] = fmaxf(pmax[qt][r], acc[qt][t][r]);
            }
#pragma unroll
    for (int qt = 0; qt < 2; ++qt)
#pragma unroll
        for (int r = 0; r < 8; ++r)
#pragma unroll
            for (int m = 1; m < 16; m <<= 1)
                pmax[qt][r] = fmaxf(pmax[qt][r], __shfl_xor(pmax[qt][r], m, 32));
    if ((lane & 15) == 0) {
#pragma unroll
        for (int qt = 0; qt < 2; ++qt)
#pragma unroll
            for (int r = 0; r < 8; ++r)
                red[wave * 32 + qt * 16 + half * 8 + r] = pmax[qt][r];
    }
    __syncthreads();
    if (threadIdx.x < 32) {                              // rowid = threadIdx.x
        float m = red[threadIdx.x];
#pragma unroll
        for (int w = 1; w < 16; ++w) m = fmaxf(m, red[w * 32 + threadIdx.x]);
        red2[threadIdx.x] = m;
    }
    __syncthreads();
    float rowmax[2][8];
#pragma unroll
    for (int qt = 0; qt < 2; ++qt)
#pragma unroll
        for (int r = 0; r < 8; ++r)
            rowmax[qt][r] = red2[qt * 16 + half * 8 + r];

    float psum[2][8];
#pragma unroll
    for (int qt = 0; qt < 2; ++qt)
#pragma unroll
        for (int r = 0; r < 8; ++r) psum[qt][r] = 0.f;
#pragma unroll
    for (int qt = 0; qt < 2; ++qt)
#pragma unroll
        for (int t = 0; t < 8; ++t)
#pragma unroll
            for (int r = 0; r < 8; ++r) {
                float e = __expf(acc[qt][t][r] - rowmax[qt][r]);
                acc[qt][t][r] = e;
                psum[qt][r] += e;
            }
#pragma unroll
    for (int qt = 0; qt < 2; ++qt)
#pragma unroll
        for (int r = 0; r < 8; ++r)
#pragma unroll
            for (int m = 1; m < 16; m <<= 1)
                psum[qt][r] += __shfl_xor(psum[qt][r], m, 32);
    if ((lane & 15) == 0) {
#pragma unroll
        for (int qt = 0; qt < 2; ++qt)
#pragma unroll
            for (int r = 0; r < 8; ++r)
                red[wave * 32 + qt * 16 + half * 8 + r] = psum[qt][r];
    }
    __syncthreads();
    if (threadIdx.x < 32) {
        float s = red[threadIdx.x];
#pragma unroll
        for (int w = 1; w < 16; ++w) s += red[w * 32 + threadIdx.x];
        red2[threadIdx.x] = s;
    }
    __syncthreads();
    float rowrcp[2][8];
#pragma unroll
    for (int qt = 0; qt < 2; ++qt)
#pragma unroll
        for (int r = 0; r < 8; ++r)
            rowrcp[qt][r] = 1.0f / red2[qt * 16 + half * 8 + r];

    // Normalize + compress P to f16 in registers (frees the 128 f32 accs).
    v8h ph[2][8];
#pragma unroll
    for (int qt = 0; qt < 2; ++qt)
#pragma unroll
        for (int t = 0; t < 8; ++t)
#pragma unroll
            for (int r = 0; r < 8; ++r)
                ph[qt][t][r] = (_Float16)(acc[qt][t][r] * rowrcp[qt][r]);
    __syncthreads();                                     // red/red2 dead; LDS free

    // ---- Phase 3: O = P @ V, two 1024-column P chunks through 64KB LDS ----
    const int d0 = wave * 64;
    v8f oacc[2][4] = {};
#pragma unroll 1
    for (int c = 0; c < 2; ++c) {
        if ((jw >> 10) == c) {                           // this wave's cols in chunk
            int jl = jw & 1023;
#pragma unroll
            for (int t = 0; t < 8; ++t) {
                int col = jl + t * 16 + (lane & 15);
#pragma unroll
                for (int qt = 0; qt < 2; ++qt)
#pragma unroll
                    for (int r = 0; r < 8; ++r)
                        Psh[qt * 16 + half * 8 + r][col] = ph[qt][t][r];
            }
        }
        __syncthreads();
        const int rr = lane & 15;
        const int kb = (lane >> 4) << 3;
        for (int j0 = 0; j0 < 1024; j0 += 32) {
            union { v16h v; v8h h[2]; } ua0, ua1;
            ua0.h[0] = *(const v8h*)&Psh[rr][j0 + kb];
            ua0.h[1] = *(const v8h*)&Psh[rr][j0 + kb + 16];
            ua1.h[0] = *(const v8h*)&Psh[16 + rr][j0 + kb];
            ua1.h[1] = *(const v8h*)&Psh[16 + rr][j0 + kb + 16];
#pragma unroll
            for (int t = 0; t < 4; t += 2) {
                v16h bf0 = load_b16(Vb + (size_t)(d0 + t * 16) * TJ + c * 1024 + j0, TJ, lane);
                v16h bf1 = load_b16(Vb + (size_t)(d0 + (t + 1) * 16) * TJ + c * 1024 + j0, TJ, lane);
                oacc[0][t]     = WMMA16(ua0.v, bf0, oacc[0][t]);
                oacc[1][t]     = WMMA16(ua1.v, bf0, oacc[1][t]);
                oacc[0][t + 1] = WMMA16(ua0.v, bf1, oacc[0][t + 1]);
                oacc[1][t + 1] = WMMA16(ua1.v, bf1, oacc[1][t + 1]);
            }
        }
        __syncthreads();                                 // before chunk overwrite
    }

    float* Ob = O + ((size_t)batch * TI + q0) * DIM;
#pragma unroll
    for (int t = 0; t < 4; ++t) {
        int col = d0 + t * 16 + (lane & 15);
#pragma unroll
        for (int qt = 0; qt < 2; ++qt) {
            int rbase = qt * 16 + half * 8;
#pragma unroll
            for (int r = 0; r < 8; ++r)
                Ob[(size_t)(rbase + r) * DIM + col] = oacc[qt][t][r];
        }
    }
}

// ---------------------------------------------------------------------------
// Host-side launch
// ---------------------------------------------------------------------------
extern "C" void kernel_launch(void* const* d_in, const int* in_sizes, int n_in,
                              void* d_out, int out_size, void* d_ws, size_t ws_size,
                              hipStream_t stream) {
    (void)in_sizes; (void)n_in; (void)out_size; (void)ws_size;

    const float* target   = (const float*)d_in[0];
    const float* source_k = (const float*)d_in[1];
    const float* source_v = (const float*)d_in[2];
    const float* Wq = (const float*)d_in[3];
    const float* bq = (const float*)d_in[4];
    const float* Wk = (const float*)d_in[5];
    const float* bk = (const float*)d_in[6];
    const float* Wv = (const float*)d_in[7];
    const float* bv = (const float*)d_in[8];
    const float* g_t = (const float*)d_in[9];
    const float* b_t = (const float*)d_in[10];
    const float* g_k = (const float*)d_in[11];
    const float* b_k = (const float*)d_in[12];
    const float* g_v = (const float*)d_in[13];
    const float* b_v = (const float*)d_in[14];
    float* out = (float*)d_out;

    // Workspace layout (f16 activations + f16 weights), ~198 MB total
    const size_t NE = (size_t)BQ * TI * DIM;            // 16 Mi elements
    char* ws = (char*)d_ws;
    _Float16* tn  = (_Float16*)(ws);                    // normalized target
    _Float16* skn = (_Float16*)(ws + 2 * NE);           // normalized source_k
    _Float16* svn = (_Float16*)(ws + 4 * NE);           // normalized source_v
    _Float16* Qh  = (_Float16*)(ws + 6 * NE);           // Q f16 (16384,1024)
    _Float16* Kh  = (_Float16*)(ws + 8 * NE);           // K f16 (16384,1024)
    _Float16* Vth = (_Float16*)(ws + 10 * NE);          // Vt f16 (8,1024,2048)
    _Float16* Whq = (_Float16*)(ws + 12 * NE);
    _Float16* Whk = (_Float16*)(ws + 12 * NE + 2u * DIM * DIM);
    _Float16* Whv = (_Float16*)(ws + 12 * NE + 4u * DIM * DIM);

    const int rows = BQ * TI;                           // 16384

    // 1) LayerNorm + cast
    ln_cast_kernel<<<rows, 256, 0, stream>>>(target,   g_t, b_t, tn);
    ln_cast_kernel<<<rows, 256, 0, stream>>>(source_k, g_k, b_k, skn);
    ln_cast_kernel<<<rows, 256, 0, stream>>>(source_v, g_v, b_v, svn);

    // 2) Weight conversion
    const int wn = DIM * DIM;
    f32_to_f16_kernel<<<wn / 256, 256, 0, stream>>>(Wq, Whq, wn);
    f32_to_f16_kernel<<<wn / 256, 256, 0, stream>>>(Wk, Whk, wn);
    f32_to_f16_kernel<<<wn / 256, 256, 0, stream>>>(Wv, Whv, wn);

    // 3) Projections (WMMA): Q, K row-major; V transposed per batch
    proj_gemm_kernel<<<rows / 32, 256, 0, stream>>>(tn,  Whq, bq, Qh,  0);
    proj_gemm_kernel<<<rows / 32, 256, 0, stream>>>(skn, Whk, bk, Kh,  0);
    proj_gemm_kernel<<<rows / 32, 256, 0, stream>>>(svn, Whv, bv, Vth, 1);

    // 4) Attention (WMMA flash-style, exact softmax, chunked P through LDS)
    attn_kernel<<<BQ * (TI / 32), 512, 0, stream>>>(Qh, Kh, Vth, out);
}
